// Net_42820823941601
// MI455X (gfx1250) — matrix operand order repaired and checked by reading, not compiled, not run on hardware
//
#include <hip/hip_runtime.h>

typedef __attribute__((ext_vector_type(16))) __bf16 v16bf;
typedef __attribute__((ext_vector_type(8)))  float  v8f;

#define WAVES_PER_BLOCK 4

// ---------------------------------------------------------------------------
// Weight pre-pack: convert W[k][f][h] (fp32) into per-lane B fragments (bf16)
// matching the 32x16 bf16 WMMA layout:
//   lane (hi,loc): element j holds B[K = s*32 + 8*hi + j (j<8) | s*32 + 16 + 8*hi + (j-8) (j>=8)][N = ct*16 + loc]
// Linear layout: (((k*col_tiles + ct)*ksteps + s)*32 + lane)*16 + j
// ---------------------------------------------------------------------------
__global__ void pack_w_bf16(const float* __restrict__ W, __bf16* __restrict__ Wpk,
                            int fin, int fout, int K, int col_tiles, int ksteps)
{
    int i = blockIdx.x * blockDim.x + threadIdx.x;
    int total = K * col_tiles * ksteps * 32 * 16;
    if (i >= total) return;
    int j    = i & 15;
    int lane = (i >> 4) & 31;
    int f    = i >> 9;                  // fragment index
    int s    = f % ksteps;
    int ct   = (f / ksteps) % col_tiles;
    int k    = (f / ksteps) / col_tiles;
    int hi   = lane >> 4, loc = lane & 15;
    int kk   = s * 32 + (j < 8 ? 8 * hi + j : 16 + 8 * hi + (j - 8));
    int n_g  = ct * 16 + loc;
    float v  = (kk < fin && n_g < fout) ? W[((long)k * fin + kk) * fout + n_g] : 0.0f;
    Wpk[i] = (__bf16)v;
}

// ---------------------------------------------------------------------------
// WMMA GEMM:  Y[n, k, h] = sum_f X[n, k, f] * W[k, f, h]
//   X element (n,k,f) = X[n*x_row_stride + k*x_plane_stride + f]   (fp32)
//   B fragments pre-packed bf16 (see pack_w_bf16)
//   Y element (n,k,h) = Y[n*y_row_stride + k*y_plane_stride + h]
// grid: (ceil(rowtiles/4), col_tiles, K), block 128 (4 waves),
// one wave per 16x16 output tile, v_wmma_f32_16x16x32_bf16.
// FIN is compile time; all loads are unconditional (clamped), masking via
// cndmask-selects only -> EXEC untouched around WMMA.
// ---------------------------------------------------------------------------
template <int FIN>
__global__ void gemm_bf16_wmma(const float* __restrict__ X,
                               const __bf16* __restrict__ Wpk,
                               float* __restrict__ Y,
                               int nrows, int fout,
                               int x_row_stride, int x_plane_stride,
                               int y_row_stride, int y_plane_stride,
                               int col_tiles)
{
    constexpr int KSTEPS = (FIN + 31) / 32;
    const int lane     = threadIdx.x & 31;
    const int wave     = threadIdx.x >> 5;
    const int row_tile = blockIdx.x * WAVES_PER_BLOCK + wave;
    const int col_tile = blockIdx.y;
    const int k        = blockIdx.z;
    const int row0     = row_tile * 16;
    if (row0 >= nrows) return;               // wave-uniform early out

    const int loc = lane & 15;
    const int hi  = (lane >> 4) & 1;
    int m = row0 + loc;
    if (m >= nrows) m = nrows - 1;           // clamp: garbage lands only in masked D rows
    const int n_g = col_tile * 16 + loc;

    const float* xrow = X + (long)m * x_row_stride + (long)k * x_plane_stride;
    const __bf16* wp  = Wpk + ((((long)k * col_tiles + col_tile) * KSTEPS) * 32 + lane) * 16;

    v8f c = {};
    #pragma unroll
    for (int s = 0; s < KSTEPS; ++s) {
        v16bf a;
        // ---- A fragment: 16x32 bf16 layout.
        // group0: K = s*32 + 8*hi + (0..7); group1: K = s*32 + 16 + 8*hi + (0..7)
        {
            const int g0 = s * 32 + 8 * hi;
            const bool v0 = (g0 + 8 <= FIN);
            const float4* p = (const float4*)(xrow + (v0 ? g0 : 0));
            float4 f0 = p[0], f1 = p[1];
            a[0] = v0 ? (__bf16)f0.x : (__bf16)0.0f;
            a[1] = v0 ? (__bf16)f0.y : (__bf16)0.0f;
            a[2] = v0 ? (__bf16)f0.z : (__bf16)0.0f;
            a[3] = v0 ? (__bf16)f0.w : (__bf16)0.0f;
            a[4] = v0 ? (__bf16)f1.x : (__bf16)0.0f;
            a[5] = v0 ? (__bf16)f1.y : (__bf16)0.0f;
            a[6] = v0 ? (__bf16)f1.z : (__bf16)0.0f;
            a[7] = v0 ? (__bf16)f1.w : (__bf16)0.0f;
        }
        {
            const int g1 = s * 32 + 16 + 8 * hi;
            const bool v1 = (g1 + 8 <= FIN);
            const float4* p = (const float4*)(xrow + (v1 ? g1 : 0));
            float4 f0 = p[0], f1 = p[1];
            a[8]  = v1 ? (__bf16)f0.x : (__bf16)0.0f;
            a[9]  = v1 ? (__bf16)f0.y : (__bf16)0.0f;
            a[10] = v1 ? (__bf16)f0.z : (__bf16)0.0f;
            a[11] = v1 ? (__bf16)f0.w : (__bf16)0.0f;
            a[12] = v1 ? (__bf16)f1.x : (__bf16)0.0f;
            a[13] = v1 ? (__bf16)f1.y : (__bf16)0.0f;
            a[14] = v1 ? (__bf16)f1.z : (__bf16)0.0f;
            a[15] = v1 ? (__bf16)f1.w : (__bf16)0.0f;
        }
        // ---- B fragment: one aligned 32-byte vector load
        v16bf b = *(const v16bf*)(wp + (long)s * 512);

        c = __builtin_amdgcn_wmma_f32_16x16x32_bf16(
                false, a, false, b, (short)0, c, false, false);
    }
    // C/D layout: c[r] = D[M = r + 8*hi][N = loc]
    if (n_g < fout) {
        #pragma unroll
        for (int r = 0; r < 8; ++r) {
            int rr = row0 + r + 8 * hi;
            if (rr < nrows)
                Y[(long)rr * y_row_stride + (long)k * y_plane_stride + n_g] = c[r];
        }
    }
}

// ---------------------------------------------------------------------------
// Graph kernels
// ---------------------------------------------------------------------------
__global__ void zero_f(float* __restrict__ p, int n)
{
    int i = blockIdx.x * blockDim.x + threadIdx.x;
    if (i < n) p[i] = 0.0f;
}

__global__ void degree_kernel(const int* __restrict__ col, float* __restrict__ deg, int E)
{
    int e = blockIdx.x * blockDim.x + threadIdx.x;
    if (e < E)
        __hip_atomic_fetch_add(&deg[col[e]], 1.0f,
                               __ATOMIC_RELAXED, __HIP_MEMORY_SCOPE_AGENT);
}

__global__ void dinv_kernel(float* __restrict__ deg, int n)
{
    int i = blockIdx.x * blockDim.x + threadIdx.x;
    if (i < n) {
        float d = deg[i];
        deg[i] = (d > 0.0f) ? rsqrtf(d) : 0.0f;
    }
}

__global__ void ew_kernel(const int* __restrict__ row, const int* __restrict__ col,
                          const float* __restrict__ dinv, float* __restrict__ ew, int E)
{
    int e = blockIdx.x * blockDim.x + threadIdx.x;
    if (e < E) ew[e] = dinv[row[e]] * dinv[col[e]];
}

// scatter: dst[col[e]] += src[row[e]] * ew[e], KF floats per node (vec2/lane).
template <int KF>
__global__ void scatter_kernel(const float* __restrict__ src, float* __restrict__ dst,
                               const int* __restrict__ row, const int* __restrict__ col,
                               const float* __restrict__ ew, int E)
{
    constexpr int CH = KF / 2;
    long t = (long)blockIdx.x * blockDim.x + threadIdx.x;
    long total = (long)E * CH;
    if (t >= total) return;
    int e = (int)(t / CH);
    int c = (int)(t % CH) * 2;
    float w = ew[e];
    const float* s = src + (long)row[e] * KF + c;
    float*       d = dst + (long)col[e] * KF + c;
    __hip_atomic_fetch_add(&d[0], s[0] * w, __ATOMIC_RELAXED, __HIP_MEMORY_SCOPE_AGENT);
    __hip_atomic_fetch_add(&d[1], s[1] * w, __ATOMIC_RELAXED, __HIP_MEMORY_SCOPE_AGENT);
}

// dst = dst + root + bias[kf]; optional relu.  bias is [K*FoutK] = KF contiguous.
__global__ void combine_kernel(float* __restrict__ dst, const float* __restrict__ root,
                               const float* __restrict__ bias, int total, int KF, int relu)
{
    int i = blockIdx.x * blockDim.x + threadIdx.x;
    if (i >= total) return;
    int kf = i % KF;
    float v = dst[i] + root[i] + bias[kf];
    if (relu) v = fmaxf(v, 0.0f);
    dst[i] = v;
}

// h[n,f] = relu( 0.5*(q[n, 0, f] + q[n, 1, f]) )
__global__ void mean2_relu_kernel(const float* __restrict__ q, float* __restrict__ h,
                                  int N, int F, int KF)
{
    int i = blockIdx.x * blockDim.x + threadIdx.x;
    if (i >= N * F) return;
    int n = i / F, f = i % F;
    float v = 0.5f * (q[(long)n * KF + f] + q[(long)n * KF + F + f]);
    h[i] = fmaxf(v, 0.0f);
}

// out[n,:] = log_softmax( 0.5*(q[n,0,:] + q[n,1,:]) )
template <int F>
__global__ void mean2_logsoftmax_kernel(const float* __restrict__ q, float* __restrict__ out,
                                        int N, int KF)
{
    int n = blockIdx.x * blockDim.x + threadIdx.x;
    if (n >= N) return;
    float vals[F];
    float mx = -3.0e38f;
    #pragma unroll
    for (int f = 0; f < F; ++f) {
        float v = 0.5f * (q[(long)n * KF + f] + q[(long)n * KF + F + f]);
        vals[f] = v;
        mx = fmaxf(mx, v);
    }
    float s = 0.0f;
    #pragma unroll
    for (int f = 0; f < F; ++f) s += expf(vals[f] - mx);
    float lse = logf(s) + mx;
    #pragma unroll
    for (int f = 0; f < F; ++f) out[(long)n * F + f] = vals[f] - lse;
}

// ---------------------------------------------------------------------------
// Host launch
// ---------------------------------------------------------------------------
static inline int cdiv(long a, long b) { return (int)((a + b - 1) / b); }

extern "C" void kernel_launch(void* const* d_in, const int* in_sizes, int n_in,
                              void* d_out, int out_size, void* d_ws, size_t ws_size,
                              hipStream_t stream)
{
    const int NUM_IN = 128, NUM_HID = 32, NUM_OUT = 40, K = 2;
    const int N = in_sizes[0] / NUM_IN;   // 50000
    const int E = in_sizes[1] / 2;        // 800000

    const float* x       = (const float*)d_in[0];
    const int*   eidx    = (const int*)d_in[1];
    const int*   row     = eidx;          // edge_index[0]
    const int*   col     = eidx + E;      // edge_index[1]
    const float* w1_init = (const float*)d_in[2];
    const float* w1_deep = (const float*)d_in[3];
    const float* w1_root = (const float*)d_in[4];
    const float* b1      = (const float*)d_in[5];
    const float* w2_init = (const float*)d_in[6];
    const float* w2_deep = (const float*)d_in[7];
    const float* w2_root = (const float*)d_in[8];
    const float* b2      = (const float*)d_in[9];
    float*       out     = (float*)d_out;

    // ---- workspace bump allocation (256B aligned) ----
    char* wp = (char*)d_ws;
    auto alloc = [&](size_t bytes) -> void* {
        void* p = (void*)wp;
        wp += (bytes + 255) & ~(size_t)255;
        return p;
    };
    const int KF1 = K * NUM_HID;   // 64
    const int KF2 = K * NUM_OUT;   // 80
    float* deg   = (float*)alloc((size_t)N * 4);
    float* ew    = (float*)alloc((size_t)E * 4);
    float* P1    = (float*)alloc((size_t)N * KF1 * 4);
    float* Q1    = (float*)alloc((size_t)N * KF1 * 4);
    float* root1 = (float*)alloc((size_t)N * KF1 * 4);
    float* h1    = (float*)alloc((size_t)N * NUM_HID * 4);
    float* P2    = (float*)alloc((size_t)N * KF2 * 4);
    float* Q2    = (float*)alloc((size_t)N * KF2 * 4);
    float* root2 = (float*)alloc((size_t)N * KF2 * 4);

    // packed bf16 B fragments (tiny)
    const int CT1 = cdiv(NUM_HID, 16);   // 2
    const int CT2 = cdiv(NUM_OUT, 16);   // 3
    const int KS128 = 4, KS32 = 1, KS40 = 2;
    auto pkbytes = [&](int ct, int ks) { return (size_t)K * ct * ks * 512 * 2; };
    __bf16* pk1i = (__bf16*)alloc(pkbytes(CT1, KS128));
    __bf16* pk1r = (__bf16*)alloc(pkbytes(CT1, KS128));
    __bf16* pk1d = (__bf16*)alloc(pkbytes(CT1, KS32));
    __bf16* pk2i = (__bf16*)alloc(pkbytes(CT2, KS32));
    __bf16* pk2r = (__bf16*)alloc(pkbytes(CT2, KS32));
    __bf16* pk2d = (__bf16*)alloc(pkbytes(CT2, KS40));

    const int TB = 256;
    const int row_tiles = cdiv(N, 16);
    const int gx = cdiv(row_tiles, WAVES_PER_BLOCK);
    dim3 blk(32 * WAVES_PER_BLOCK);

    // ---- pack weights ----
    auto pack = [&](const float* W, __bf16* dst, int fin, int fout, int ct, int ks) {
        int total = K * ct * ks * 512;
        pack_w_bf16<<<cdiv(total, TB), TB, 0, stream>>>(W, dst, fin, fout, K, ct, ks);
    };
    pack(w1_init, pk1i, NUM_IN,  NUM_HID, CT1, KS128);
    pack(w1_root, pk1r, NUM_IN,  NUM_HID, CT1, KS128);
    pack(w1_deep, pk1d, NUM_HID, NUM_HID, CT1, KS32);
    pack(w2_init, pk2i, NUM_HID, NUM_OUT, CT2, KS32);
    pack(w2_root, pk2r, NUM_HID, NUM_OUT, CT2, KS32);
    pack(w2_deep, pk2d, NUM_OUT, NUM_OUT, CT2, KS40);

    // ---- edge weights (symmetric GCN norm) ----
    zero_f<<<cdiv(N, TB), TB, 0, stream>>>(deg, N);
    degree_kernel<<<cdiv(E, TB), TB, 0, stream>>>(col, deg, E);
    dinv_kernel<<<cdiv(N, TB), TB, 0, stream>>>(deg, N);
    ew_kernel<<<cdiv(E, TB), TB, 0, stream>>>(row, col, deg, ew, E);

    // ---- conv1: x[N,128] -> Q1[N,2,32] ----
    {
        dim3 g(gx, CT1, K);
        gemm_bf16_wmma<128><<<g, blk, 0, stream>>>(x, pk1r, root1, N, NUM_HID,
                                                   NUM_IN, 0, KF1, NUM_HID, CT1);
        gemm_bf16_wmma<128><<<g, blk, 0, stream>>>(x, pk1i, P1, N, NUM_HID,
                                                   NUM_IN, 0, KF1, NUM_HID, CT1);
        // t = 0
        zero_f<<<cdiv((long)N * KF1, TB), TB, 0, stream>>>(Q1, N * KF1);
        scatter_kernel<64><<<cdiv((long)E * (KF1 / 2), TB), TB, 0, stream>>>(P1, Q1, row, col, ew, E);
        combine_kernel<<<cdiv((long)N * KF1, TB), TB, 0, stream>>>(Q1, root1, b1, N * KF1, KF1, 1);
        // t = 1: per-stack deep GEMM, then propagate again
        gemm_bf16_wmma<32><<<g, blk, 0, stream>>>(Q1, pk1d, P1, N, NUM_HID,
                                                  KF1, NUM_HID, KF1, NUM_HID, CT1);
        zero_f<<<cdiv((long)N * KF1, TB), TB, 0, stream>>>(Q1, N * KF1);
        scatter_kernel<64><<<cdiv((long)E * (KF1 / 2), TB), TB, 0, stream>>>(P1, Q1, row, col, ew, E);
        combine_kernel<<<cdiv((long)N * KF1, TB), TB, 0, stream>>>(Q1, root1, b1, N * KF1, KF1, 1);
    }
    // h1 = relu(mean_k Q1)
    mean2_relu_kernel<<<cdiv((long)N * NUM_HID, TB), TB, 0, stream>>>(Q1, h1, N, NUM_HID, KF1);

    // ---- conv2: h1[N,32] -> Q2[N,2,40] ----
    {
        dim3 g(gx, CT2, K);
        gemm_bf16_wmma<32><<<g, blk, 0, stream>>>(h1, pk2r, root2, N, NUM_OUT,
                                                  NUM_HID, 0, KF2, NUM_OUT, CT2);
        gemm_bf16_wmma<32><<<g, blk, 0, stream>>>(h1, pk2i, P2, N, NUM_OUT,
                                                  NUM_HID, 0, KF2, NUM_OUT, CT2);
        // t = 0
        zero_f<<<cdiv((long)N * KF2, TB), TB, 0, stream>>>(Q2, N * KF2);
        scatter_kernel<80><<<cdiv((long)E * (KF2 / 2), TB), TB, 0, stream>>>(P2, Q2, row, col, ew, E);
        combine_kernel<<<cdiv((long)N * KF2, TB), TB, 0, stream>>>(Q2, root2, b2, N * KF2, KF2, 0);
        // t = 1
        gemm_bf16_wmma<40><<<g, blk, 0, stream>>>(Q2, pk2d, P2, N, NUM_OUT,
                                                  KF2, NUM_OUT, KF2, NUM_OUT, CT2);
        zero_f<<<cdiv((long)N * KF2, TB), TB, 0, stream>>>(Q2, N * KF2);
        scatter_kernel<80><<<cdiv((long)E * (KF2 / 2), TB), TB, 0, stream>>>(P2, Q2, row, col, ew, E);
        combine_kernel<<<cdiv((long)N * KF2, TB), TB, 0, stream>>>(Q2, root2, b2, N * KF2, KF2, 0);
    }

    // ---- epilogue: mean over stacks + log_softmax ----
    mean2_logsoftmax_kernel<40><<<cdiv(N, TB), TB, 0, stream>>>(Q2, out, N, KF2);
    (void)n_in; (void)out_size; (void)ws_size;
}
